// DispCorrM_38998303048193
// MI455X (gfx1250) — matrix operand order repaired and checked by reading, not compiled, not run on hardware
//
#include <hip/hip_runtime.h>

// WMMA operand/result vector types (wave32, gfx1250)
typedef __attribute__((ext_vector_type(16))) __bf16 v16bf;
typedef __attribute__((ext_vector_type(8)))  float  v8f;
typedef __attribute__((ext_vector_type(4)))  float  v4f;   // native vector for NT loads

namespace {
constexpr int kB = 4, kC = 64, kH = 256, kW = 512, kD = 48;
constexpr unsigned kChanStride = (unsigned)kH * kW;       // elements between channels
// ---- LDS layout (bytes), 160KB total -> 2 workgroups per 320KB WGP ----
// Rb: right operand, WMMA-B swizzled. tile = [kh(2)][lane(32)][32B], j = 0..31 (no halo:
//     G tiles with j<0 are entirely zero and are skipped at compute time)
constexpr unsigned RB_OFF  = 0;
constexpr unsigned RB_TILE = 2048;
// La: left operand, WMMA-A swizzled, same tile geometry
constexpr unsigned LA_OFF  = RB_OFF + 32 * RB_TILE;       // 65536
constexpr unsigned LA_TILE = 2048;
// Gs: per-wave D-tile scratch: 8 waves * 4 tiles * 16*16*4B
constexpr unsigned GS_OFF  = LA_OFF + 32 * LA_TILE;       // 131072
constexpr unsigned SMEM_TOTAL = GS_OFF + 8u * 4u * 1024u; // 163840 = 160KB
}

__device__ __forceinline__ unsigned short f2bf(float f) { // f32 -> bf16 RNE
  unsigned u = __float_as_uint(f);
  u += 0x7FFFu + ((u >> 16) & 1u);
  return (unsigned short)(u >> 16);
}
__device__ __forceinline__ unsigned pk2(float a, float b) { // pack 2 bf16 (a = low addr)
  return (unsigned)f2bf(a) | ((unsigned)f2bf(b) << 16);
}

__global__ __launch_bounds__(256, 4) void disp_corr_wmma(
    const float* __restrict__ x, float* __restrict__ out) {
  extern __shared__ __align__(16) char smem[];
  const int t  = threadIdx.x;
  const int bh = blockIdx.x;
  const int b  = bh >> 8;                 // kH == 256
  const int hh = bh & (kH - 1);

  // ---- stage LEFT: x[b][0:64][hh][:] -> La, bf16, WMMA-A layout ----
  // A 16x32 (MxK), this kernel's contiguous k order (not ISA table order): lane l<16 holds M=l, K={0..7}(g0),{8..15}(g1);
  // lane l>=16 holds M=l-16, K={16..23}(g0),{24..31}(g1); 8 K-values = 16B contiguous.
  {
    const int q  = t >> 5;                // channel octet 0..7
    const int s  = t & 31;
    const int kh = q >> 2, g = (q >> 1) & 1, hi = q & 1;
    const int cbase = kh * 32 + hi * 16 + g * 8;   // c = cbase + o, o in [0,8)
    const v4f* x4 = (const v4f*)x;
    const size_t base4 = ((size_t)(b * 2 * kC + cbase) * kH + hh) * (kW / 4);
    for (int it = 0; it < 4; ++it) {
      const int wb = it * 128 + s * 4;    // lanes cover contiguous w -> coalesced b128 loads
      v4f f[8];
#pragma unroll
      for (int o = 0; o < 8; ++o)
        f[o] = __builtin_nontemporal_load(
            &x4[base4 + (size_t)o * (kChanStride / 4) + (wb >> 2)]);
#pragma unroll
      for (int dw = 0; dw < 4; ++dw) {
        const int w = wb + dw;
        uint4 pkv;
        pkv.x = pk2(f[0][dw], f[1][dw]);
        pkv.y = pk2(f[2][dw], f[3][dw]);
        pkv.z = pk2(f[4][dw], f[5][dw]);
        pkv.w = pk2(f[6][dw], f[7][dw]);
        *(uint4*)(smem + LA_OFF + (unsigned)(w >> 4) * LA_TILE + kh * 1024 +
                  (hi * 16 + (w & 15)) * 32 + g * 16) = pkv;
      }
    }
  }

  // ---- stage RIGHT: x[b][64:128][hh][:] -> Rb, bf16, WMMA-B layout ----
  // B 32x16 (KxN), contiguous k as A: lane l<16 holds N=l, K=0..15; lane>=16 holds N=l-16, K=16..31;
  // 16 K-values contiguous (32B), K pairs packed low/high per dword.
  {
    const int p  = t >> 6;                // channel 16-block 0..3
    const int s  = t & 63;
    const int kh = p >> 1, hi = p & 1;
    const int cbase = kC + p * 16;        // global channel = 64 + 16p + o
    const v4f* x4 = (const v4f*)x;
    const size_t base4 = ((size_t)(b * 2 * kC + cbase) * kH + hh) * (kW / 4);
    for (int it = 0; it < 2; ++it) {
      const int wb = it * 256 + s * 4;
      v4f f[16];
#pragma unroll
      for (int o = 0; o < 16; ++o)
        f[o] = __builtin_nontemporal_load(
            &x4[base4 + (size_t)o * (kChanStride / 4) + (wb >> 2)]);
#pragma unroll
      for (int dw = 0; dw < 4; ++dw) {
        const int w = wb + dw;
        char* dst = smem + RB_OFF + (unsigned)(w >> 4) * RB_TILE +
                    kh * 1024 + (hi * 16 + (w & 15)) * 32;
        uint4 lo, hiq;
        lo.x  = pk2(f[0][dw],  f[1][dw]);  lo.y  = pk2(f[2][dw],  f[3][dw]);
        lo.z  = pk2(f[4][dw],  f[5][dw]);  lo.w  = pk2(f[6][dw],  f[7][dw]);
        hiq.x = pk2(f[8][dw],  f[9][dw]);  hiq.y = pk2(f[10][dw], f[11][dw]);
        hiq.z = pk2(f[12][dw], f[13][dw]); hiq.w = pk2(f[14][dw], f[15][dw]);
        *(uint4*)(dst)      = lo;
        *(uint4*)(dst + 16) = hiq;
      }
    }
  }

  __syncthreads();

  // ---- banded G = L^T R via WMMA; stitch diagonals out[d,w] = G[w, w-d]/64 ----
  const int wv = t >> 5, lane = t & 31;
  float* gsl = (float*)(smem + GS_OFF + wv * 4096);
  const int Mb = (lane >= 16) ? 8 : 0;   // D layout: VGPR r, lanes 0-15: M=r; 16-31: M=r+8
  const int Nc = lane & 15;
  for (int i = wv; i < 32; i += 8) {     // i = w tile, wave-private
    const v16bf a0 = *(const v16bf*)(smem + LA_OFF + (unsigned)i * LA_TILE + lane * 32);
    const v16bf a1 = *(const v16bf*)(smem + LA_OFF + (unsigned)i * LA_TILE + 1024 + lane * 32);
#pragma unroll
    for (int off = 0; off < 4; ++off) {
      const int j = __builtin_amdgcn_readfirstlane(i + off - 3);  // w' tile (wave-uniform)
      v8f acc = {};
      if (j >= 0) {                       // j<0 tiles are entirely w'<0 -> zero (padding)
        const v16bf b0 = *(const v16bf*)(smem + RB_OFF + (unsigned)j * RB_TILE + lane * 32);
        const v16bf b1 = *(const v16bf*)(smem + RB_OFF + (unsigned)j * RB_TILE + 1024 + lane * 32);
        acc = __builtin_amdgcn_wmma_f32_16x16x32_bf16(false, a0, false, b0,
                                                      (short)0, acc, false, false);
        acc = __builtin_amdgcn_wmma_f32_16x16x32_bf16(false, a1, false, b1,
                                                      (short)0, acc, false, false);
      }
      float* gt = gsl + off * 256;
#pragma unroll
      for (int r = 0; r < 8; ++r) gt[((Mb + r) << 4) | Nc] = acc[r];
    }
    // wave-local cross-lane visibility of the D-tile scratch (LDS in-order per wave)
    asm volatile("s_wait_dscnt 0x0" ::: "memory");
    // extraction: lane handles column w0+m; d = (lane>>4) + 2*it; rel = m - d + 48 in [1,63]
    const int w0 = i << 4;
    const int m  = lane & 15;
    const int d0 = lane >> 4;
    int rel = m - d0 + 48;
    size_t oidx = ((size_t)(b * kD + d0) * kH + hh) * kW + w0 + m;
#pragma unroll
    for (int it = 0; it < 24; ++it) {
      const float v = gsl[((rel >> 4) << 8) + (m << 4) + (rel & 15)];
      __builtin_nontemporal_store(v * (1.0f / 64.0f), &out[oidx]);
      rel  -= 2;
      oidx += (size_t)2 * kH * kW;
    }
  }
}

extern "C" void kernel_launch(void* const* d_in, const int* in_sizes, int n_in,
                              void* d_out, int out_size, void* d_ws, size_t ws_size,
                              hipStream_t stream) {
  (void)in_sizes; (void)n_in; (void)out_size; (void)d_ws; (void)ws_size;
  const float* x = (const float*)d_in[0];     // (4,128,256,512) f32; d_in[1]=max_disp (48, hardcoded)
  float* out = (float*)d_out;                 // (4,48,256,512) f32
  (void)hipFuncSetAttribute((const void*)disp_corr_wmma,
                            hipFuncAttributeMaxDynamicSharedMemorySize,
                            (int)SMEM_TOTAL);
  disp_corr_wmma<<<dim3(kB * kH), dim3(256), SMEM_TOTAL, stream>>>(x, out);
}
